// VectorQuantizer_14388140441657
// MI455X (gfx1250) — matrix operand order repaired and checked by reading, not compiled, not run on hardware
//
#include <hip/hip_runtime.h>
#include <climits>
#include <cstdint>

// ---------------------------------------------------------------------------
// VectorQuantizer (Sinkhorn VQ) for MI455X / gfx1250, wave32.
//   d = |x|^2 + |e|^2 - 2 x e^T      (WMMA bf16 hi/lo split => ~fp32 accuracy;
//                                     codebook tiles staged to LDS via TDM)
//   center d to [-1,1], log-Sinkhorn 100 iters tracked as r[i], c[j] offsets
//   indices = argmax_j (s_ij + c_j);  x_q = e[idx];  loss = 1.25 * MSE
// ---------------------------------------------------------------------------

typedef __attribute__((ext_vector_type(16))) __bf16        v16bf;
typedef __attribute__((ext_vector_type(8)))  float         v8f;
typedef __attribute__((ext_vector_type(4)))  unsigned int  v4u;
typedef __attribute__((ext_vector_type(8)))  int           v8i;
typedef __attribute__((ext_vector_type(4)))  int           v4i;

constexpr int   NTOK = 65536;
constexpr int   NE   = 1024;
constexpr int   ED   = 256;
constexpr float EPS  = 0.003f;
constexpr int   ITERS = 100;
constexpr float LOGK = 6.93147180559945f;    // log(1024)
constexpr float LOGB = 11.0903548889591f;    // log(65536)
constexpr int   BSTRIDE = ED + 8;            // padded LDS row stride (elements)

// workspace layout (bytes); total ~338 MB
constexpr size_t S_OFF  = 0;                             // f32 [NTOK][NE]
constexpr size_t XH_OFF = S_OFF  + (size_t)NTOK*NE*4;    // bf16 [NTOK][ED] x hi
constexpr size_t XL_OFF = XH_OFF + (size_t)NTOK*ED*2;    // bf16 x lo
constexpr size_t EH_OFF = XL_OFF + (size_t)NTOK*ED*2;    // bf16 [NE][ED] e hi
constexpr size_t EL_OFF = EH_OFF + (size_t)NE*ED*2;      // bf16 e lo
constexpr size_t XN_OFF = EL_OFF + (size_t)NE*ED*2;      // f32 [NTOK] |x|^2
constexpr size_t EN_OFF = XN_OFF + (size_t)NTOK*4;       // f32 [NE]   |e|^2
constexpr size_t CV_OFF = EN_OFF + (size_t)NE*4;         // f32 [NE]   col offsets c
constexpr size_t RV_OFF = CV_OFF + (size_t)NE*4;         // f32 [NTOK] row offsets r
constexpr size_t PC_OFF = RV_OFF + (size_t)NTOK*4;       // float2 [64][NE] col LSE partials
constexpr size_t MM_OFF = PC_OFF + (size_t)64*NE*8;      // int[2] min/max keys
constexpr size_t LP_OFF = MM_OFF + 256;                  // f32 [8192] loss partials

#if defined(__has_builtin)
#  if __has_builtin(__builtin_amdgcn_tensor_load_to_lds)
#    define HAVE_TDM 1
#  endif
#endif
#ifndef HAVE_TDM
#  define HAVE_TDM 0
#endif

__device__ __forceinline__ void wait_tensorcnt0() {
#if defined(__has_builtin)
#  if __has_builtin(__builtin_amdgcn_s_wait_tensorcnt)
  __builtin_amdgcn_s_wait_tensorcnt(0);
  return;
#  else
  asm volatile("s_wait_tensorcnt 0x0" ::: "memory");
  return;
#  endif
#else
  asm volatile("s_wait_tensorcnt 0x0" ::: "memory");
#endif
}

#if HAVE_TDM
// Issue one 2D TDM tile load: tile_rows x row_elems (bf16) from a row-major
// tensor (tensor_rows x row_elems, stride row_elems) into LDS at lds_off,
// with pad_enable inserting 4 DWORDs (8 elems) after every 128 DWORDs (one
// 512B row) => padded LDS row stride of BSTRIDE elements.
__device__ __forceinline__ void tdm_load_tile_bf16(unsigned lds_off, const void* gsrc,
                                                   unsigned tile_rows, unsigned row_elems,
                                                   unsigned tensor_rows) {
  unsigned long long ga = (unsigned long long)gsrc;
  v4u g0;
  g0.x = 1u;                                        // count=1 (valid), gather off
  g0.y = lds_off;                                   // lds_addr (bytes, block-relative)
  g0.z = (unsigned)(ga & 0xffffffffu);              // global_addr[31:0]
  g0.w = (unsigned)((ga >> 32) & 0x01ffffffu)       // global_addr[56:32]
       | (2u << 30);                                // type=2 ("image")
  v8i g1;
  g1[0] = (int)((1u << 16)                          // data_size=1 => 2 bytes
              | (1u << 20)                          // pad_enable
              | (6u << 22)                          // pad_interval: 128 DWORDs
              | (3u << 25));                        // pad_amount: 4 DWORDs
  g1[1] = (int)((row_elems & 0xffffu) << 16);       // tensor_dim0[15:0] @ bits63:48
  g1[2] = (int)(((row_elems >> 16) & 0xffffu)       // tensor_dim0[31:16]
              | ((tensor_rows & 0xffffu) << 16));   // tensor_dim1[15:0]
  g1[3] = (int)(((tensor_rows >> 16) & 0xffffu)     // tensor_dim1[31:16]
              | ((row_elems & 0xffffu) << 16));     // tile_dim0 @ bits127:112
  g1[4] = (int)(tile_rows & 0xffffu);               // tile_dim1 (tile_dim2 = 0)
  g1[5] = (int)row_elems;                           // tensor_dim0_stride[31:0]
  g1[6] = 0;                                        // stride0 hi | stride1 lo
  g1[7] = 0;
  v4i z4 = (v4i){0, 0, 0, 0};
#if __clang_major__ >= 23
  v8i z8 = (v8i){0, 0, 0, 0, 0, 0, 0, 0};
  __builtin_amdgcn_tensor_load_to_lds(g0, g1, z4, z4, z8, 0);
#else
  __builtin_amdgcn_tensor_load_to_lds(g0, g1, z4, z4, 0);
#endif
}
#endif

// monotone int key for floats (order-preserving)
__device__ __forceinline__ int fkey(float f) {
  int i = __float_as_int(f);
  return i >= 0 ? i : (i ^ 0x7fffffff);
}
__device__ __forceinline__ float fdec(int k) {
  return __int_as_float(k >= 0 ? k : (k ^ 0x7fffffff));
}

// streaming log-sum-exp state updates
__device__ __forceinline__ void lse_add(float& m, float& s, float v) {
  if (v <= m) { s += __expf(v - m); }
  else        { s = s * __expf(m - v) + 1.0f; m = v; }
}
__device__ __forceinline__ void lse_comb(float& m, float& s, float m2, float s2) {
  float M = fmaxf(m, m2);
  if (M == -INFINITY) { m = M; s = 0.0f; return; }
  s = s * __expf(m - M) + s2 * __expf(m2 - M);
  m = M;
}

// ---------------- init ----------------
__global__ void vq_init(int* mm, float* c) {
  if (threadIdx.x == 0) { mm[0] = INT_MAX; mm[1] = INT_MIN; }
  for (int k = threadIdx.x; k < NE; k += 256) c[k] = 0.0f;
}

// ---------------- prep: row norms + bf16 hi/lo split ----------------
__global__ __launch_bounds__(256) void vq_prep_x(const float* __restrict__ x,
                                                 __bf16* __restrict__ xh, __bf16* __restrict__ xl,
                                                 float* __restrict__ xn, float* __restrict__ r) {
  const int lane = threadIdx.x & 31, wid = threadIdx.x >> 5;
  const int row = blockIdx.x * 8 + wid;
  const float* p = x + (size_t)row * ED;
  float4 a = *(const float4*)(p + lane * 4);
  float4 b = *(const float4*)(p + 128 + lane * 4);
  float sq = a.x*a.x + a.y*a.y + a.z*a.z + a.w*a.w
           + b.x*b.x + b.y*b.y + b.z*b.z + b.w*b.w;
  const float va[8] = {a.x, a.y, a.z, a.w, b.x, b.y, b.z, b.w};
  #pragma unroll
  for (int cth = 0; cth < 8; ++cth) {
    int k = (cth < 4 ? lane * 4 + cth : 128 + lane * 4 + (cth - 4));
    float v = va[cth];
    __bf16 h = (__bf16)v;
    __bf16 l = (__bf16)(v - (float)h);
    xh[(size_t)row * ED + k] = h;
    xl[(size_t)row * ED + k] = l;
  }
  #pragma unroll
  for (int off = 16; off > 0; off >>= 1) sq += __shfl_xor(sq, off, 32);
  if (lane == 0) { xn[row] = sq; r[row] = 0.0f; }
}

__global__ __launch_bounds__(256) void vq_prep_e(const float* __restrict__ e,
                                                 __bf16* __restrict__ eh, __bf16* __restrict__ el,
                                                 float* __restrict__ en) {
  const int lane = threadIdx.x & 31, wid = threadIdx.x >> 5;
  const int row = blockIdx.x * 8 + wid;
  const float* p = e + (size_t)row * ED;
  float4 a = *(const float4*)(p + lane * 4);
  float4 b = *(const float4*)(p + 128 + lane * 4);
  float sq = a.x*a.x + a.y*a.y + a.z*a.z + a.w*a.w
           + b.x*b.x + b.y*b.y + b.z*b.z + b.w*b.w;
  const float va[8] = {a.x, a.y, a.z, a.w, b.x, b.y, b.z, b.w};
  #pragma unroll
  for (int cth = 0; cth < 8; ++cth) {
    int k = (cth < 4 ? lane * 4 + cth : 128 + lane * 4 + (cth - 4));
    float v = va[cth];
    __bf16 h = (__bf16)v;
    __bf16 l = (__bf16)(v - (float)h);
    eh[(size_t)row * ED + k] = h;
    el[(size_t)row * ED + k] = l;
  }
  #pragma unroll
  for (int off = 16; off > 0; off >>= 1) sq += __shfl_xor(sq, off, 32);
  if (lane == 0) en[row] = sq;
}

// ---------------- WMMA fragment loads (CDNA5 16x16x32 bf16 layouts) ----------------
// A 16x32: lane holds row rowBase+(lane&15); elems 0..7 = K kb+8*sel.., 8..15 = K kb+16+8*sel..
__device__ __forceinline__ v16bf ld_frag_a(const __bf16* base, int row, int kb, int sel) {
  const __bf16* p = base + (size_t)row * ED + kb + sel * 8;
  union { v16bf v; float4 q[2]; } u;
  u.q[0] = *(const float4*)(p);
  u.q[1] = *(const float4*)(p + 16);
  return u.v;
}
// B 32x16 (= e^T) from the padded LDS tile: lane holds column (local) col;
// elems 0..7 = K kb+8*sel.., 8..15 = K kb+16+8*sel.. (same as A)
__device__ __forceinline__ v16bf ld_frag_b_lds(const __bf16* base, int col, int kb, int sel) {
  const __bf16* p = base + col * BSTRIDE + kb + sel * 8;
  union { v16bf v; float4 q[2]; } u;
  u.q[0] = *(const float4*)(p);
  u.q[1] = *(const float4*)(p + 16);
  return u.v;
}
__device__ __forceinline__ v8f wmma_bf16(v16bf a, v16bf b, v8f c) {
  return __builtin_amdgcn_wmma_f32_16x16x32_bf16(false, a, false, b, (short)0, c, false, false);
}

// ---------------- distances via WMMA: d = |x|^2 + |e|^2 - 2 x.e ----------------
// block = 8 waves; wave computes 16 rows x 64 cols; grid (512, 16).
// The 64-col codebook slice (hi+lo) is staged into LDS by the Tensor Data Mover.
// Two accumulator banks (hh vs hl+lh) keep same-acc WMMA RAW deps >= 4 apart.
__global__ __launch_bounds__(256) void vq_dist_wmma(const float* __restrict__ xn, const float* __restrict__ en,
                                                    const __bf16* __restrict__ xh, const __bf16* __restrict__ xl,
                                                    const __bf16* __restrict__ eh, const __bf16* __restrict__ el,
                                                    float* __restrict__ S, int* __restrict__ mm) {
  __shared__ __align__(16) __bf16 btile[2][64 * BSTRIDE];  // ~66KB, hi + lo
  __shared__ int smin[8], smax[8];
  const int lane = threadIdx.x & 31, wid = threadIdx.x >> 5;
  const int sel = lane >> 4, lr = lane & 15;
  const int rowBase = blockIdx.x * 128 + wid * 16;
  const int colBase = blockIdx.y * 64;

#if HAVE_TDM
  // keep a (never-taken, runtime-unprovable) visible writer of btile so the
  // optimizer cannot undef-fold the LDS reads that the TDM actually produces
  if ((int)blockIdx.x == -1) {
    btile[0][threadIdx.x] = (__bf16)1.0f;
    btile[1][threadIdx.x] = (__bf16)1.0f;
  }
  if (wid == 0) {
    tdm_load_tile_bf16(0u, eh + (size_t)colBase * ED, 64u, (unsigned)ED, (unsigned)NE);
    tdm_load_tile_bf16((unsigned)(64 * BSTRIDE * 2), el + (size_t)colBase * ED,
                       64u, (unsigned)ED, (unsigned)NE);
    wait_tensorcnt0();
  }
  // compiler fence: TDM wrote LDS behind the optimizer's back
  asm volatile("" ::: "memory");
#else
  // fallback: cooperative staging into the same padded LDS layout
  for (int t = threadIdx.x; t < 64 * (ED / 8); t += 256) {
    int rrow = t >> 5, ch = t & 31;
    *(float4*)(&btile[0][rrow * BSTRIDE + ch * 8]) =
        *(const float4*)(eh + (size_t)(colBase + rrow) * ED + ch * 8);
    *(float4*)(&btile[1][rrow * BSTRIDE + ch * 8]) =
        *(const float4*)(el + (size_t)(colBase + rrow) * ED + ch * 8);
  }
#endif
  __syncthreads();

  v8f accA[4], accB[4];
  #pragma unroll
  for (int t = 0; t < 4; ++t) {
    accA[t] = (v8f){0.f,0.f,0.f,0.f,0.f,0.f,0.f,0.f};
    accB[t] = (v8f){0.f,0.f,0.f,0.f,0.f,0.f,0.f,0.f};
  }

  #pragma unroll
  for (int kb = 0; kb < ED; kb += 32) {
    v16bf ah = ld_frag_a(xh, rowBase + lr, kb, sel);
    v16bf al = ld_frag_a(xl, rowBase + lr, kb, sel);
    v16bf bh[4], bl[4];
    #pragma unroll
    for (int ct = 0; ct < 4; ++ct) {
      bh[ct] = ld_frag_b_lds(btile[0], ct * 16 + lr, kb, sel);
      bl[ct] = ld_frag_b_lds(btile[1], ct * 16 + lr, kb, sel);
    }
    #pragma unroll
    for (int ct = 0; ct < 4; ++ct) accA[ct] = wmma_bf16(ah, bh[ct], accA[ct]);  // hi*hi
    #pragma unroll
    for (int ct = 0; ct < 4; ++ct) accB[ct] = wmma_bf16(ah, bl[ct], accB[ct]);  // hi*lo
    #pragma unroll
    for (int ct = 0; ct < 4; ++ct) accB[ct] = wmma_bf16(al, bh[ct], accB[ct]);  // lo*hi
  }

  int kmin = INT_MAX, kmax = INT_MIN;
  #pragma unroll
  for (int ct = 0; ct < 4; ++ct) {
    const int n = colBase + ct * 16 + lr;
    const float env = en[n];
    #pragma unroll
    for (int rr = 0; rr < 8; ++rr) {
      const int m = rowBase + rr + sel * 8;     // D layout: VGPR rr, half-wave sel
      float d = xn[m] + env - 2.0f * (accA[ct][rr] + accB[ct][rr]);
      S[(size_t)m * NE + n] = d;
      int kk = fkey(d);
      kmin = min(kmin, kk);
      kmax = max(kmax, kk);
    }
  }
  #pragma unroll
  for (int off = 16; off > 0; off >>= 1) {
    kmin = min(kmin, __shfl_xor(kmin, off, 32));
    kmax = max(kmax, __shfl_xor(kmax, off, 32));
  }
  if (lane == 0) { smin[wid] = kmin; smax[wid] = kmax; }
  __syncthreads();
  if (threadIdx.x == 0) {
    int a = smin[0], b = smax[0];
    #pragma unroll
    for (int w = 1; w < 8; ++w) { a = min(a, smin[w]); b = max(b, smax[w]); }
    atomicMin(&mm[0], a);
    atomicMax(&mm[1], b);
  }
}

// ---------------- center + scale into log-space scores: s = (mid - d)/(amp*eps) ----------------
__global__ __launch_bounds__(256) void vq_scale(float* __restrict__ S, const int* __restrict__ mm) {
  const float mx = fdec(mm[1]), mn = fdec(mm[0]);
  const float mid = 0.5f * (mx + mn);
  const float amp = mx - mid + 1e-5f;
  const float sc = 1.0f / (amp * EPS);
  size_t i = (size_t)blockIdx.x * 256 + threadIdx.x;   // one float4 per thread
  float4* p = (float4*)S;
  float4 v = p[i];
  v.x = (mid - v.x) * sc;
  v.y = (mid - v.y) * sc;
  v.z = (mid - v.z) * sc;
  v.w = (mid - v.w) * sc;
  p[i] = v;
}

// ---------------- Sinkhorn column pass, stage 1: partial LSE over 1024-row chunks ----------------
// grid (8, 64): x = 128-col group, y = 1024-row chunk. threads: tx=col(128), ty=row phase(2)
__global__ __launch_bounds__(256) void vq_col_partial(const float* __restrict__ S,
                                                      const float* __restrict__ r,
                                                      float2* __restrict__ pc) {
  __shared__ float sm[128], ss[128];
  const int tx = threadIdx.x & 127, ty = threadIdx.x >> 7;
  const int col = blockIdx.x * 128 + tx;
  float m = -INFINITY, s = 0.0f;
  for (int it = 0; it < 512; ++it) {
    int i = blockIdx.y * 1024 + ty + it * 2;
    float v = S[(size_t)i * NE + col] + r[i];
    lse_add(m, s, v);
  }
  if (ty == 1) { sm[tx] = m; ss[tx] = s; }
  __syncthreads();
  if (ty == 0) {
    lse_comb(m, s, sm[tx], ss[tx]);
    pc[(size_t)blockIdx.y * NE + col] = make_float2(m, s);
  }
}

// stage 2: combine 64 partials per column -> c_j = -LSE_i(s+r) - logK
__global__ __launch_bounds__(256) void vq_col_combine(const float2* __restrict__ pc,
                                                      float* __restrict__ c) {
  const int col = blockIdx.x * 256 + threadIdx.x;
  float m = -INFINITY, s = 0.0f;
  for (int ch = 0; ch < 64; ++ch) {
    float2 p = pc[(size_t)ch * NE + col];
    lse_comb(m, s, p.x, p.y);
  }
  c[col] = -(m + __logf(s)) - LOGK;
}

// ---------------- Sinkhorn row pass: r_i = -LSE_j(s+c) - logB (one wave32 per row) ----------------
__global__ __launch_bounds__(256) void vq_row_update(const float* __restrict__ S,
                                                     const float* __restrict__ c,
                                                     float* __restrict__ r) {
  __shared__ float cs[NE];
  for (int k = threadIdx.x; k < NE; k += 256) cs[k] = c[k];
  __syncthreads();
  const int lane = threadIdx.x & 31, wid = threadIdx.x >> 5;
  const int row = blockIdx.x * 8 + wid;
  const float* srow = S + (size_t)row * NE;
  float m = -INFINITY, s = 0.0f;
  #pragma unroll 4
  for (int jj = 0; jj < 32; ++jj) {
    int j = lane + jj * 32;
    if (jj + 8 < 32) __builtin_prefetch(srow + j + 8 * 32, 0, 1);  // global_prefetch_b8
    lse_add(m, s, srow[j] + cs[j]);
  }
  #pragma unroll
  for (int off = 16; off > 0; off >>= 1) {
    float om = __shfl_xor(m, off, 32);
    float os = __shfl_xor(s, off, 32);
    lse_comb(m, s, om, os);
  }
  if (lane == 0) r[row] = -(m + __logf(s)) - LOGB;
}

// ---------------- final: argmax, codebook gather, loss partials ----------------
__global__ __launch_bounds__(256) void vq_final(const float* __restrict__ S,
                                                const float* __restrict__ c,
                                                const float* __restrict__ e,
                                                const float* __restrict__ x,
                                                float* __restrict__ outx,
                                                float* __restrict__ outidx,
                                                float* __restrict__ lossp) {
  __shared__ float cs[NE];
  __shared__ float werr[8];
  for (int k = threadIdx.x; k < NE; k += 256) cs[k] = c[k];
  __syncthreads();
  const int lane = threadIdx.x & 31, wid = threadIdx.x >> 5;
  const int row = blockIdx.x * 8 + wid;
  const float* srow = S + (size_t)row * NE;

  float best = -INFINITY; int bj = NE;
  for (int jj = 0; jj < 32; ++jj) {
    int j = lane + jj * 32;
    float v = srow[j] + cs[j];
    if (v > best) { best = v; bj = j; }   // j strictly increases per lane -> first max kept
  }
  #pragma unroll
  for (int off = 16; off > 0; off >>= 1) {
    float ob = __shfl_xor(best, off, 32);
    int   oj = __shfl_xor(bj, off, 32);
    if (ob > best || (ob == best && oj < bj)) { best = ob; bj = oj; }
  }
  // all lanes now agree on argmax bj; gather codebook row + squared error
  const float* erow = e + (size_t)bj * ED;
  const float* xrow = x + (size_t)row * ED;
  float* orow = outx + (size_t)row * ED;
  float err = 0.0f;
  float4 ea = *(const float4*)(erow + lane * 4);
  float4 eb = *(const float4*)(erow + 128 + lane * 4);
  float4 xa = *(const float4*)(xrow + lane * 4);
  float4 xb = *(const float4*)(xrow + 128 + lane * 4);
  *(float4*)(orow + lane * 4) = ea;
  *(float4*)(orow + 128 + lane * 4) = eb;
  float dx;
  dx = ea.x - xa.x; err += dx * dx;  dx = ea.y - xa.y; err += dx * dx;
  dx = ea.z - xa.z; err += dx * dx;  dx = ea.w - xa.w; err += dx * dx;
  dx = eb.x - xb.x; err += dx * dx;  dx = eb.y - xb.y; err += dx * dx;
  dx = eb.z - xb.z; err += dx * dx;  dx = eb.w - xb.w; err += dx * dx;
  #pragma unroll
  for (int off = 16; off > 0; off >>= 1) err += __shfl_xor(err, off, 32);
  if (lane == 0) { werr[wid] = err; outidx[row] = (float)bj; }
  __syncthreads();
  if (threadIdx.x == 0) {
    float t = 0.0f;
    #pragma unroll
    for (int w = 0; w < 8; ++w) t += werr[w];
    lossp[blockIdx.x] = t;
  }
}

// deterministic fixed-order loss reduction: loss = (1+beta)*mean
__global__ __launch_bounds__(256) void vq_loss_reduce(const float* __restrict__ lossp,
                                                      float* __restrict__ loss) {
  __shared__ float sh[256];
  float t = 0.0f;
  for (int k = 0; k < 32; ++k) t += lossp[threadIdx.x + 256 * k];
  sh[threadIdx.x] = t;
  __syncthreads();
  for (int off = 128; off > 0; off >>= 1) {
    if (threadIdx.x < off) sh[threadIdx.x] += sh[threadIdx.x + off];
    __syncthreads();
  }
  if (threadIdx.x == 0)
    loss[0] = sh[0] * 1.25f / ((float)NTOK * (float)ED);
}

// ---------------------------------------------------------------------------
extern "C" void kernel_launch(void* const* d_in, const int* in_sizes, int n_in,
                              void* d_out, int out_size, void* d_ws, size_t ws_size,
                              hipStream_t stream) {
  (void)in_sizes; (void)n_in; (void)out_size; (void)ws_size;
  const float* x = (const float*)d_in[0];        // [65536, 256] f32
  const float* e = (const float*)d_in[1];        // [1024, 256] f32

  char* ws = (char*)d_ws;
  float*  S  = (float*)(ws + S_OFF);
  __bf16* XH = (__bf16*)(ws + XH_OFF);
  __bf16* XL = (__bf16*)(ws + XL_OFF);
  __bf16* EH = (__bf16*)(ws + EH_OFF);
  __bf16* EL = (__bf16*)(ws + EL_OFF);
  float*  XN = (float*)(ws + XN_OFF);
  float*  EN = (float*)(ws + EN_OFF);
  float*  CV = (float*)(ws + CV_OFF);
  float*  RV = (float*)(ws + RV_OFF);
  float2* PC = (float2*)(ws + PC_OFF);
  int*    MM = (int*)(ws + MM_OFF);
  float*  LP = (float*)(ws + LP_OFF);

  float* outx  = (float*)d_out;                  // x_q_out [65536*256]
  float* lossP = outx + (size_t)NTOK * ED;       // scalar loss
  float* oidx  = lossP + 1;                      // indices [65536] (as float)

  vq_init<<<1, 256, 0, stream>>>(MM, CV);
  vq_prep_x<<<NTOK / 8, 256, 0, stream>>>(x, XH, XL, XN, RV);
  vq_prep_e<<<NE / 8, 256, 0, stream>>>(e, EH, EL, EN);
  vq_dist_wmma<<<dim3(NTOK / 128, NE / 64), 256, 0, stream>>>(XN, EN, XH, XL, EH, EL, S, MM);
  vq_scale<<<(NTOK * (size_t)NE / 4) / 256, 256, 0, stream>>>(S, MM);

  for (int it = 0; it < ITERS; ++it) {
    vq_col_partial<<<dim3(NE / 128, NTOK / 1024), 256, 0, stream>>>(S, RV, PC);
    vq_col_combine<<<NE / 256, 256, 0, stream>>>(PC, CV);
    vq_row_update<<<NTOK / 8, 256, 0, stream>>>(S, CV, RV);
  }

  vq_final<<<NTOK / 8, 256, 0, stream>>>(S, CV, e, x, outx, oidx, LP);
  vq_loss_reduce<<<1, 256, 0, stream>>>(LP, lossP);
}
